// WaveNet_31645319036933
// MI455X (gfx1250) — matrix operand order repaired and checked by reading, not compiled, not run on hardware
//
#include <hip/hip_runtime.h>
#include <hip/hip_bf16.h>

typedef __attribute__((ext_vector_type(16))) __bf16 v16bf;
typedef __attribute__((ext_vector_type(8)))  __bf16 v8bf;
typedef __attribute__((ext_vector_type(8)))  float  v8f;

#define B_      8
#define T_      8192
#define HID_    128
#define INCH_   256
#define MIDC_   512
#define OUTC_   256
#define NL_     20
#define GOODC_  2046
#define TARGET_ 6146

__device__ __forceinline__ v8f wmma_bf16(v16bf a, v16bf b, v8f c) {
  // D = A(16x32) * B(32x16) + C, bf16 inputs, f32 accumulate
  return __builtin_amdgcn_wmma_f32_16x16x32_bf16(false, a, false, b, (short)0, c, false, false);
}

// CDNA5 async copy: 16B global -> LDS per lane, tracked by ASYNCcnt.
__device__ __forceinline__ void async_copy_16B(unsigned lds_byte_off, const void* gaddr) {
  asm volatile("global_load_async_to_lds_b128 %0, %1, off"
               :: "v"(lds_byte_off), "v"((unsigned long long)gaddr)
               : "memory");
}

__device__ __forceinline__ void wait_asynccnt0() {
#if __has_builtin(__builtin_amdgcn_s_wait_asynccnt)
  __builtin_amdgcn_s_wait_asynccnt(0);
#else
  asm volatile("s_wait_asynccnt 0x0" ::: "memory");
#endif
}

// ---------------------------------------------------------------------------
// Weight repack: per 16x32 A-tile, lane-major fragment order.
// Element e of lane L maps to K = e + 16*(L>>4), M = L&15.
// Layer weights: [layer][mat:5][mtile:8][ktile:4][lane:32][e:16]
// mat 0/1 = Wf tap0/tap1, 2/3 = Wg tap0/tap1, 4 = Wr
// ---------------------------------------------------------------------------
__global__ void prep_layer_weights(const float* __restrict__ Wf,
                                   const float* __restrict__ Wg,
                                   const float* __restrict__ Wr,
                                   __bf16* __restrict__ dst) {
  int flat = blockIdx.x * 256 + threadIdx.x;          // < 20*81920
  int layer = flat / 81920; int r  = flat % 81920;
  int mat   = r / 16384;    int r2 = r % 16384;
  int mtile = r2 / 2048;    int r3 = r2 % 2048;
  int ktile = r3 / 512;     int r4 = r3 % 512;
  int lane  = r4 / 16;      int e  = r4 % 16;
  int K = ktile * 32 + e + 16 * (lane >> 4);
  int M = mtile * 16 + (lane & 15);
  int base = (layer * HID_ + M) * HID_ + K;
  float v;
  if      (mat == 0) v = Wf[base * 2 + 0];
  else if (mat == 1) v = Wf[base * 2 + 1];
  else if (mat == 2) v = Wg[base * 2 + 0];
  else if (mat == 3) v = Wg[base * 2 + 1];
  else               v = Wr[base];
  dst[flat] = (__bf16)v;
}

// Generic matrix repack (row-major src, stride = K extent)
__global__ void prep_mat(const float* __restrict__ src, __bf16* __restrict__ dst,
                         int Ktiles, int rowStride) {
  int flat = blockIdx.x * 256 + threadIdx.x;
  int e    = flat & 15;
  int lane = (flat >> 4) & 31;
  int kt   = (flat >> 9) % Ktiles;
  int mt   = (flat >> 9) / Ktiles;
  int K = kt * 32 + e + 16 * (lane >> 4);
  int M = mt * 16 + (lane & 15);
  dst[flat] = (__bf16)src[(size_t)M * rowStride + K];
}

// inputs (B,256,T) fp32 channel-major -> [b][t][256] bf16 channel-last
__global__ void transpose_inputs(const float* __restrict__ in, __bf16* __restrict__ dst) {
  int flat = blockIdx.x * 256 + threadIdx.x;          // < B*T*INCH
  int c = flat & (INCH_ - 1);
  int t = (flat >> 8) & (T_ - 1);
  int b = flat >> 21;
  dst[flat] = (__bf16)in[((size_t)(b * INCH_ + c)) * T_ + t];
}

// ---------------------------------------------------------------------------
// Input 1x1 conv: res = Wi(128x256) * inp + bi. 8 waves = 8 M-tiles,
// 4 time-tiles (64 cols) per block.
// ---------------------------------------------------------------------------
__global__ void input_conv(const __bf16* __restrict__ inpb,
                           const __bf16* __restrict__ wiF,
                           const float* __restrict__ bi,
                           float* __restrict__ res, __bf16* __restrict__ resb) {
  int lane = threadIdx.x & 31, wave = threadIdx.x >> 5;
  int hi = lane >> 4, col = lane & 15;
  int b = blockIdx.y, t0 = blockIdx.x * 64;
  v8f acc[4] = {};
#pragma unroll
  for (int k = 0; k < 8; k++) {
    v16bf a = *(const v16bf*)(wiF + (wave * 8 + k) * 512 + lane * 16);
#pragma unroll
    for (int i = 0; i < 4; i++) {
      int t = t0 + i * 16 + col;
      v16bf bb = *(const v16bf*)(inpb + (size_t)(b * T_ + t) * INCH_ + k * 32 + hi * 16);
      acc[i] = wmma_bf16(a, bb, acc[i]);
    }
  }
  int chb = wave * 16 + hi * 8;
  float bias[8];
#pragma unroll
  for (int r = 0; r < 8; r++) bias[r] = bi[chb + r];
#pragma unroll
  for (int i = 0; i < 4; i++) {
    int t = t0 + i * 16 + col;
    size_t o = (size_t)(b * T_ + t) * HID_ + chb;
    v8f xr; v8bf xb;
#pragma unroll
    for (int r = 0; r < 8; r++) { float x = acc[i][r] + bias[r]; xr[r] = x; xb[r] = (__bf16)x; }
    *(v8f*)(res + o)   = xr;
    *(v8bf*)(resb + o) = xb;
  }
}

// ---------------------------------------------------------------------------
// One WaveNet residual layer. Absolute-column indexing: new res valid on
// [off_new, T). y[a] = Wf0*res[a-d] + Wf1*res[a]; x = Wr*(tanh*sigmoid);
// res_new[a] = x + res_old[a]; final[a-2046] += x for a >= 2046.
// Block = 8 waves (one M-tile each) x 4 time tiles (64 cols). The two dilated
// taps are contiguous 64-col spans: staged once per block into LDS via
// global_load_async_to_lds_b128 (ASYNCcnt), then all 8 waves read B fragments
// from LDS (was 8x redundant global traffic). Gated output staged through a
// second LDS tile as the B matrix of the Wr GEMM.
// ---------------------------------------------------------------------------
__global__ void layer_kernel(const __bf16* __restrict__ resb_in,
                             const float*  __restrict__ res_in,
                             __bf16* __restrict__ resb_out,
                             float*  __restrict__ res_out,
                             float*  __restrict__ finalb,
                             const __bf16* __restrict__ wl,
                             const float* __restrict__ bf,
                             const float* __restrict__ bg,
                             const float* __restrict__ br,
                             int d, int off_new, int first) {
  // tap spans: [2][64 cols][144 bf16] (288B padded rows, 32B-aligned)
  __shared__ v16bf spanv[2 * 64 * 9];
  // gated output: [4 tiles][16 cols][144 bf16]
  __shared__ v16bf ldsv[4 * 16 * 9];
  __bf16* sb0 = (__bf16*)spanv;            // tap0 (column a-d)
  __bf16* sb1 = (__bf16*)spanv + 64 * 144; // tap1 (column a)
  __bf16* lds = (__bf16*)ldsv;

  int tid = threadIdx.x;
  int lane = tid & 31, m = tid >> 5;
  int hi = lane >> 4, col = lane & 15;
  int b = blockIdx.y;
  int tstart = off_new & ~15;
  int t0 = tstart + blockIdx.x * 64;

  // ---- async stage: 2 spans x 64 cols x 256B = 2 x 16KB, 16B per lane ----
#pragma unroll
  for (int j = 0; j < 4; j++) {
    int flat  = j * 256 + tid;             // 0..1023
    int colr  = flat >> 4;                 // 0..63
    int chunk = flat & 15;                 // 16B chunk within 256B row
    int g1 = t0 + colr;      g1 = g1 < 0 ? 0 : (g1 > T_ - 1 ? T_ - 1 : g1);
    int g0 = t0 - d + colr;  g0 = g0 < 0 ? 0 : (g0 > T_ - 1 ? T_ - 1 : g0);
    const char* a1 = (const char*)(resb_in + (size_t)(b * T_ + g1) * HID_) + chunk * 16;
    const char* a0 = (const char*)(resb_in + (size_t)(b * T_ + g0) * HID_) + chunk * 16;
    async_copy_16B((unsigned)(uintptr_t)((char*)sb1 + colr * 288 + chunk * 16), a1);
    async_copy_16B((unsigned)(uintptr_t)((char*)sb0 + colr * 288 + chunk * 16), a0);
  }
  wait_asynccnt0();
  __syncthreads();

  const __bf16* WF0 = wl + (0 * 8 + m) * 2048;
  const __bf16* WF1 = wl + (1 * 8 + m) * 2048;
  const __bf16* WG0 = wl + (2 * 8 + m) * 2048;
  const __bf16* WG1 = wl + (3 * 8 + m) * 2048;
  const __bf16* WR  = wl + (4 * 8 + m) * 2048;

  v8f accF[4] = {}, accG[4] = {};
#pragma unroll
  for (int k = 0; k < 4; k++) {
    v16bf af0 = *(const v16bf*)(WF0 + k * 512 + lane * 16);
    v16bf af1 = *(const v16bf*)(WF1 + k * 512 + lane * 16);
    v16bf ag0 = *(const v16bf*)(WG0 + k * 512 + lane * 16);
    v16bf ag1 = *(const v16bf*)(WG1 + k * 512 + lane * 16);
#pragma unroll
    for (int i = 0; i < 4; i++) {
      int sc = i * 16 + col;               // span column 0..63
      v16bf b0 = *(const v16bf*)(sb0 + sc * 144 + k * 32 + hi * 16);
      v16bf b1 = *(const v16bf*)(sb1 + sc * 144 + k * 32 + hi * 16);
      accF[i] = wmma_bf16(af0, b0, accF[i]);
      accF[i] = wmma_bf16(af1, b1, accF[i]);
      accG[i] = wmma_bf16(ag0, b0, accG[i]);
      accG[i] = wmma_bf16(ag1, b1, accG[i]);
    }
  }
  int chb = m * 16 + hi * 8;
  float bfv[8], bgv[8], brv[8];
#pragma unroll
  for (int r = 0; r < 8; r++) { bfv[r] = bf[chb + r]; bgv[r] = bg[chb + r]; brv[r] = br[chb + r]; }

#pragma unroll
  for (int i = 0; i < 4; i++) {
    v8bf o;
#pragma unroll
    for (int r = 0; r < 8; r++) {
      float f = tanhf(accF[i][r] + bfv[r]);
      float g = 1.f / (1.f + __expf(-(accG[i][r] + bgv[r])));
      o[r] = (__bf16)(f * g);
    }
    *(v8bf*)&lds[(i * 16 + col) * 144 + chb] = o;
  }
  __syncthreads();

  v8f accX[4] = {};
#pragma unroll
  for (int k = 0; k < 4; k++) {
    v16bf ar = *(const v16bf*)(WR + k * 512 + lane * 16);
#pragma unroll
    for (int i = 0; i < 4; i++) {
      v16bf bb = *(const v16bf*)&lds[(i * 16 + col) * 144 + k * 32 + hi * 16];
      accX[i] = wmma_bf16(ar, bb, accX[i]);
    }
  }

#pragma unroll
  for (int i = 0; i < 4; i++) {
    int a = t0 + i * 16 + col;
    if (a < off_new || a >= T_) continue;
    size_t o = (size_t)(b * T_ + a) * HID_ + chb;
    v8f resid = *(const v8f*)(res_in + o);
    v8f xr, nr; v8bf nb;
#pragma unroll
    for (int r = 0; r < 8; r++) {
      float x = accX[i][r] + brv[r];
      xr[r] = x;
      float n = x + resid[r];
      nr[r] = n; nb[r] = (__bf16)n;
    }
    *(v8f*)(res_out + o)   = nr;
    *(v8bf*)(resb_out + o) = nb;
    if (a >= GOODC_) {
      size_t fo = (size_t)(b * TARGET_ + (a - GOODC_)) * HID_ + chb;
      if (first) {
        *(v8f*)(finalb + fo) = xr;
      } else {
        v8f fv = *(const v8f*)(finalb + fo);
#pragma unroll
        for (int r = 0; r < 8; r++) fv[r] += xr[r];
        *(v8f*)(finalb + fo) = fv;
      }
    }
  }
}

// ---------------------------------------------------------------------------
// Tail: mid = relu(W1(512x128)*relu(final) + b1), out = W2(256x512)*mid + b2
// ---------------------------------------------------------------------------
__global__ void tail1(const float* __restrict__ finalb,
                      const __bf16* __restrict__ w1F,
                      const float* __restrict__ b1,
                      __bf16* __restrict__ mid) {
  int lane = threadIdx.x & 31, wave = threadIdx.x >> 5;
  int hi = lane >> 4, col = lane & 15;
  int b = blockIdx.y;
  int t  = blockIdx.x * 16 + col;
  int tc = t < TARGET_ ? t : TARGET_ - 1;
  v16bf bfr[4];
#pragma unroll
  for (int k = 0; k < 4; k++) {
    const float* fp = finalb + (size_t)(b * TARGET_ + tc) * HID_ + k * 32 + hi * 16;
    v16bf tmp;
#pragma unroll
    for (int j = 0; j < 16; j++) { float v = fp[j]; tmp[j] = (__bf16)(v > 0.f ? v : 0.f); }
    bfr[k] = tmp;
  }
#pragma unroll
  for (int ms = 0; ms < 4; ms++) {
    int mt = wave + ms * 8;
    v8f acc = {};
#pragma unroll
    for (int k = 0; k < 4; k++) {
      v16bf a = *(const v16bf*)(w1F + (mt * 4 + k) * 512 + lane * 16);
      acc = wmma_bf16(a, bfr[k], acc);
    }
    if (t < TARGET_) {
      int chb = mt * 16 + hi * 8;
      v8bf o;
#pragma unroll
      for (int r = 0; r < 8; r++) { float v = acc[r] + b1[chb + r]; o[r] = (__bf16)(v > 0.f ? v : 0.f); }
      *(v8bf*)(mid + (size_t)(b * TARGET_ + t) * MIDC_ + chb) = o;
    }
  }
}

__global__ void tail2(const __bf16* __restrict__ mid,
                      const __bf16* __restrict__ w2F,
                      const float* __restrict__ b2,
                      float* __restrict__ outp) {
  int lane = threadIdx.x & 31, wave = threadIdx.x >> 5;
  int hi = lane >> 4, col = lane & 15;
  int b = blockIdx.y;
  int t  = blockIdx.x * 16 + col;
  int tc = t < TARGET_ ? t : TARGET_ - 1;
  v8f acc0 = {}, acc1 = {};
#pragma unroll
  for (int k = 0; k < 16; k++) {
    v16bf bb = *(const v16bf*)(mid + (size_t)(b * TARGET_ + tc) * MIDC_ + k * 32 + hi * 16);
    v16bf a0 = *(const v16bf*)(w2F + (wave * 16 + k) * 512 + lane * 16);
    v16bf a1 = *(const v16bf*)(w2F + ((wave + 8) * 16 + k) * 512 + lane * 16);
    acc0 = wmma_bf16(a0, bb, acc0);
    acc1 = wmma_bf16(a1, bb, acc1);
  }
  if (t < TARGET_) {
    size_t ro = (size_t)(b * TARGET_ + t) * OUTC_;
    int chb0 = wave * 16 + hi * 8, chb1 = (wave + 8) * 16 + hi * 8;
    v8f o0, o1;
#pragma unroll
    for (int r = 0; r < 8; r++) { o0[r] = acc0[r] + b2[chb0 + r]; o1[r] = acc1[r] + b2[chb1 + r]; }
    *(v8f*)(outp + ro + chb0) = o0;
    *(v8f*)(outp + ro + chb1) = o1;
  }
}

// ---------------------------------------------------------------------------
extern "C" void kernel_launch(void* const* d_in, const int* in_sizes, int n_in,
                              void* d_out, int out_size, void* d_ws, size_t ws_size,
                              hipStream_t stream) {
  (void)in_sizes; (void)n_in; (void)out_size; (void)ws_size;
  const float* inputs = (const float*)d_in[0];
  const float* Wi = (const float*)d_in[1];
  const float* bi = (const float*)d_in[2];
  const float* Wf = (const float*)d_in[3];
  const float* bf = (const float*)d_in[4];
  const float* Wg = (const float*)d_in[5];
  const float* bg = (const float*)d_in[6];
  const float* Wr = (const float*)d_in[7];
  const float* br = (const float*)d_in[8];
  const float* W1 = (const float*)d_in[9];
  const float* b1 = (const float*)d_in[10];
  const float* W2 = (const float*)d_in[11];
  const float* b2 = (const float*)d_in[12];
  float* outp = (float*)d_out;

  char* base = (char*)d_ws;
  size_t off = 0;
  auto alloc = [&](size_t bytes) -> char* {
    char* p = base + off;
    off = (off + bytes + 255) & ~(size_t)255;
    return p;
  };
  float*  res0   = (float*)alloc((size_t)B_ * T_ * HID_ * 4);
  float*  res1   = (float*)alloc((size_t)B_ * T_ * HID_ * 4);
  __bf16* resb0  = (__bf16*)alloc((size_t)B_ * T_ * HID_ * 2);
  __bf16* resb1  = (__bf16*)alloc((size_t)B_ * T_ * HID_ * 2);
  float*  finalb = (float*)alloc((size_t)B_ * TARGET_ * HID_ * 4);
  char*   unbuf  = alloc((size_t)B_ * TARGET_ * MIDC_ * 2);   // >= B*T*INCH*2; reused
  __bf16* inpb   = (__bf16*)unbuf;   // phase 1
  __bf16* mid    = (__bf16*)unbuf;   // phase 3 (disjoint in time)
  __bf16* wl     = (__bf16*)alloc((size_t)NL_ * 81920 * 2);
  __bf16* wiF    = (__bf16*)alloc((size_t)32768 * 2);
  __bf16* w1F    = (__bf16*)alloc((size_t)65536 * 2);
  __bf16* w2F    = (__bf16*)alloc((size_t)131072 * 2);

  prep_layer_weights<<<6400, 256, 0, stream>>>(Wf, Wg, Wr, wl);
  prep_mat<<<128, 256, 0, stream>>>(Wi, wiF, 8, 256);
  prep_mat<<<256, 256, 0, stream>>>(W1, w1F, 4, 128);
  prep_mat<<<512, 256, 0, stream>>>(W2, w2F, 16, 512);
  transpose_inputs<<<(B_ * T_ * INCH_) / 256, 256, 0, stream>>>(inputs, inpb);
  input_conv<<<dim3(T_ / 64, B_), 256, 0, stream>>>(inpb, wiF, bi, res0, resb0);

  int offc = 0, ping = 0;
  for (int l = 0; l < NL_; l++) {
    int dd = 1 << (l % 10);
    offc += dd;
    int tstart = offc & ~15;
    int ntiles = (T_ - tstart) / 16;
    int groups = (ntiles + 3) / 4;
    const float*  rin   = ping ? res1  : res0;
    float*        rout  = ping ? res0  : res1;
    const __bf16* rbin  = ping ? resb1 : resb0;
    __bf16*       rbout = ping ? resb0 : resb1;
    layer_kernel<<<dim3(groups, B_), 256, 0, stream>>>(
        rbin, rin, rbout, rout, finalb,
        wl + (size_t)l * 81920, bf + l * HID_, bg + l * HID_, br + l * HID_,
        dd, offc, l == 0 ? 1 : 0);
    ping ^= 1;
  }

  int ttiles = (TARGET_ + 15) / 16;   // 385
  tail1<<<dim3(ttiles, B_), 256, 0, stream>>>(finalb, w1F, b1, mid);
  tail2<<<dim3(ttiles, B_), 256, 0, stream>>>(mid, w2F, b2, outp);
}
